// GraphRelationLayer_52475910423184
// MI455X (gfx1250) — matrix operand order repaired and checked by reading, not compiled, not run on hardware
//
#include <hip/hip_runtime.h>
#include <math.h>

// ---------------------------------------------------------------------------
// GraphRelationLayer for MI455X (gfx1250): bf16 WMMA pipeline, v2.
// Shapes: BS=32, N_NODE=200, H=100, N_HEAD=34, K=2, SENT_DIM=1024.
// All hot-loop operands are pre-padded bf16 so fragment loads are
// unconditional 16B/32B vector loads (no divergent guards).
// ---------------------------------------------------------------------------

typedef __attribute__((ext_vector_type(16))) __bf16 v16bf;
typedef __attribute__((ext_vector_type(8)))  __bf16 v8bf;
typedef __attribute__((ext_vector_type(8)))  float  v8f;

#define BSZ 32
#define NN  200
#define HH  100
#define RR  34
#define SD  1024
#define HP  112    // padded H (+ D channel at col 100)
#define NPAD 208   // padded N for row dimension
#define NP  224    // padded N for K dimension of stage 2
#define KP  128    // padded H for K dimension of stage 1

__device__ __forceinline__ v8f wmma_bf16(v16bf a, v16bf b, v8f c) {
  return __builtin_amdgcn_wmma_f32_16x16x32_bf16(
      false, a, false, b, (short)0, c, false, false);
}

// ---------------------------------------------------------------------------
// Padding / conversion pre-kernels (elementwise, run once per call).
// ---------------------------------------------------------------------------

// Xp[b][m(208)][c(128)] = (m<200 && c<100) ? X[b][m][c]*start[b][m] : 0
__global__ __launch_bounds__(256)
void xpad_kernel(const float* __restrict__ X, const float* __restrict__ start_attn,
                 unsigned short* __restrict__ Xp_us)
{
  __bf16* Xp = (__bf16*)Xp_us;
  const int id = blockIdx.x*256 + threadIdx.x;
  if (id >= BSZ*NPAD*KP) return;
  const int c = id & (KP-1);
  const int t = id >> 7;
  const int m = t % NPAD;
  const int b = t / NPAD;
  float v = 0.0f;
  if (m < NN && c < HH) v = X[((size_t)b*NN + m)*HH + c] * start_attn[b*NN + m];
  Xp[id] = (__bf16)v;
}

// Wp[t][r][e(112)][d(128)] = (e<100 && d<100) ? w_vs[t][r][d][e] : 0  (transposed)
__global__ __launch_bounds__(256)
void wpad_kernel(const float* __restrict__ w_vs, unsigned short* __restrict__ Wp_us)
{
  __bf16* Wp = (__bf16*)Wp_us;
  const int id = blockIdx.x*256 + threadIdx.x;
  if (id >= 2*RR*HP*KP) return;
  const int d = id & (KP-1);
  const int t2 = id >> 7;
  const int e  = t2 % HP;
  const int tr = t2 / HP;          // t*RR + r
  const int r  = tr % RR;
  const int t  = tr / RR;
  float v = 0.0f;
  if (e < HH && d < HH) v = w_vs[((size_t)(t*(RR+1) + r)*HH + d)*HH + e];
  Wp[id] = (__bf16)v;
}

// Ap[br][i(208)][j(224)] = (i<200 && j<200) ? A[br][i][j] : 0  (bf16, exact for 0/1)
__global__ __launch_bounds__(256)
void apad_kernel(const float* __restrict__ A, unsigned short* __restrict__ Ap_us)
{
  __bf16* Ap = (__bf16*)Ap_us;
  const int id = blockIdx.x*256 + threadIdx.x;    // one thread = 8 j's
  const int total = BSZ*RR*NPAD*(NP/8);
  if (id >= total) return;
  const int j0 = (id % (NP/8))*8;
  const int t  = id / (NP/8);
  const int i  = t % NPAD;
  const int br = t / NPAD;
  v8bf o;
  if (i < NN && j0 + 7 < NN) {
    const float* ar = A + ((size_t)br*NN + i)*NN + j0;
#pragma unroll
    for (int k = 0; k < 8; ++k) o[k] = (__bf16)ar[k];
  } else if (i < NN && j0 < NN) {
    const float* ar = A + ((size_t)br*NN + i)*NN;
#pragma unroll
    for (int k = 0; k < 8; ++k) o[k] = (__bf16)((j0 + k < NN) ? ar[j0 + k] : 0.0f);
  } else {
#pragma unroll
    for (int k = 0; k < 8; ++k) o[k] = (__bf16)0.0f;
  }
  *(v8bf*)(Ap + ((size_t)br*NPAD + i)*NP + j0) = o;
}

// ---------------------------------------------------------------------------
// Stage kernel: one workgroup (8 waves) per (b, r).
//   Stage 1: P = (In @ W[T,r]) * uni  -> PT[d(112)][j(224)] in LDS (transposed),
//            PT[100][*] carries the degree vector * uni.
//   Stage 2: Q = A[b,r] @ P -> Zh[b][r][d(112)][n(208)] bf16.
// A-fragments are loaded once per (mt,kk) and applied to 7 n-tile accumulators.
// ---------------------------------------------------------------------------
template<int T>
__global__ __launch_bounds__(256)
void stage_kernel(const unsigned short* __restrict__ Xp_us,
                  const unsigned short* __restrict__ Ap_us,
                  const unsigned short* __restrict__ Wp_us,
                  const float* __restrict__ start_attn,
                  const float* __restrict__ uni_attn,
                  const unsigned short* __restrict__ Zmix_us,
                  unsigned short* __restrict__ Zh_us)
{
  const int wg   = blockIdx.x;
  const int b    = wg / RR, r = wg % RR;
  const int tid  = threadIdx.x;
  const int wave = tid >> 5;
  const int l    = tid & 31;
  const int g    = l >> 4;     // lane half (K-group select)
  const int lr   = l & 15;     // row (A) / col (B,C/D) within tile

  const __bf16* Xp   = (const __bf16*)Xp_us;
  const __bf16* Ap   = (const __bf16*)Ap_us;
  const __bf16* Wp   = (const __bf16*)Wp_us;
  const __bf16* Zmix = (const __bf16*)Zmix_us;
  __bf16* Zh = (__bf16*)Zh_us;

  __shared__ alignas(32) __bf16 PT[HP][NP];   // 50,176 B

  // zero PT (pad columns must be 0.0, never stale/NaN)
  {
    v8bf z;
#pragma unroll
    for (int i = 0; i < 8; ++i) z[i] = (__bf16)0.0f;
    v8bf* p = (v8bf*)&PT[0][0];
    for (int i = tid; i < HP*NP/8; i += 256) p[i] = z;
  }
  __syncthreads();

  const float u = uni_attn[b*RR + r];
  const __bf16* WpR = Wp + (size_t)(T*RR + r)*HP*KP;   // Wp[T][r][e][d]

  // ---- Stage 1: In(208x128) @ W(128x112) ----
  for (int mt = wave; mt < 13; mt += 8) {
    v8f acc[7];
#pragma unroll
    for (int nt = 0; nt < 7; ++nt)
#pragma unroll
      for (int v = 0; v < 8; ++v) acc[nt][v] = 0.0f;

    const int m = mt*16 + lr;
    const __bf16* inrow = (T == 0)
        ? (Xp + ((size_t)b*NPAD + m)*KP)
        : (Zmix + (((size_t)b*RR + r)*NPAD + m)*KP);
#pragma unroll
    for (int kk = 0; kk < 4; ++kk) {
      const int cbase = kk*32 + g*16;
      const v8bf s0 = *(const v8bf*)(inrow + cbase);
      const v8bf s1 = *(const v8bf*)(inrow + cbase + 8);
      v16bf af;
#pragma unroll
      for (int ii = 0; ii < 8; ++ii) { af[ii] = s0[ii]; af[ii+8] = s1[ii]; }
#pragma unroll
      for (int nt = 0; nt < 7; ++nt) {
        const v16bf bf = *(const v16bf*)(WpR + (size_t)(nt*16 + lr)*KP + kk*32 + g*16);
        acc[nt] = wmma_bf16(af, bf, acc[nt]);
      }
    }
#pragma unroll
    for (int nt = 0; nt < 7; ++nt) {
      const int d  = nt*16 + lr;
      const int jb = mt*16 + g*8;    // C/D layout: M = v + 8*g
      v8bf o;
#pragma unroll
      for (int v = 0; v < 8; ++v) o[v] = (__bf16)(acc[nt][v]*u);
      *(v8bf*)&PT[d][jb] = o;
    }
  }
  __syncthreads();

  // ---- D channel row: PT[100][j] = dvec[j] * uni ----
  for (int j = tid; j < NP; j += 256) {
    float dv = 0.0f;
    if (j < NN) {
      if (T == 0) dv = start_attn[b*NN + j] * u;
      else        dv = (float)Zmix[(((size_t)b*RR + r)*NPAD + j)*KP + 100] * u;
    }
    PT[100][j] = (__bf16)dv;
  }
  __syncthreads();

  // ---- Stage 2: A(208x224) @ P(224x112) ----
  const __bf16* ApR = Ap + (size_t)(b*RR + r)*NPAD*NP;
  for (int mt = wave; mt < 13; mt += 8) {
    v8f acc[7];
#pragma unroll
    for (int nt = 0; nt < 7; ++nt)
#pragma unroll
      for (int v = 0; v < 8; ++v) acc[nt][v] = 0.0f;

    const __bf16* arow = ApR + (size_t)(mt*16 + lr)*NP;
#pragma unroll
    for (int kk = 0; kk < 7; ++kk) {
      const int jb = kk*32 + g*16;
      const v8bf s0 = *(const v8bf*)(arow + jb);
      const v8bf s1 = *(const v8bf*)(arow + jb + 8);
      v16bf af;
#pragma unroll
      for (int ii = 0; ii < 8; ++ii) { af[ii] = s0[ii]; af[ii+8] = s1[ii]; }
#pragma unroll
      for (int nt = 0; nt < 7; ++nt) {
        const v16bf bf = *(const v16bf*)&PT[nt*16 + lr][kk*32 + g*16];
        acc[nt] = wmma_bf16(af, bf, acc[nt]);
      }
    }
#pragma unroll
    for (int nt = 0; nt < 7; ++nt) {
      const int dcol = nt*16 + lr;
      const int nb   = mt*16 + g*8;
      v8bf o;
#pragma unroll
      for (int v = 0; v < 8; ++v) o[v] = (__bf16)acc[nt][v];
      *(v8bf*)(Zh + (((size_t)b*RR + r)*HP + dcol)*NPAD + nb) = o;
    }
  }
}

// ---------------------------------------------------------------------------
// Head mixing: Zmix[b][s][n][dk] = sum_r trans[b,r,s] * Zh[b][r][dk][n]
// (rows n>=200 and cols dk>100 zero-padded).
// ---------------------------------------------------------------------------
__global__ __launch_bounds__(256)
void mix_kernel(const unsigned short* __restrict__ Zh_us,
                const float* __restrict__ trans,
                unsigned short* __restrict__ Zmix_us)
{
  const __bf16* Zh = (const __bf16*)Zh_us;
  __bf16* Zmix = (__bf16*)Zmix_us;
  const int id = blockIdx.x*256 + threadIdx.x;
  if (id >= BSZ*NPAD*KP) return;
  const int dk = id & (KP-1);
  const int t2 = id >> 7;
  const int n  = t2 % NPAD;
  const int b  = t2 / NPAD;
  if (dk > HH || n >= NN) {
    for (int s = 0; s < RR; ++s)
      Zmix[(((size_t)b*RR + s)*NPAD + n)*KP + dk] = (__bf16)0.0f;
    return;
  }
  float vals[RR];
#pragma unroll
  for (int rr = 0; rr < RR; ++rr)
    vals[rr] = (float)Zh[(((size_t)b*RR + rr)*HP + dk)*NPAD + n];
  const float* tb = trans + (size_t)b*RR*RR;
  for (int s = 0; s < RR; ++s) {
    float acc = 0.0f;
#pragma unroll
    for (int rr = 0; rr < RR; ++rr) acc += vals[rr]*tb[rr*RR + s];
    Zmix[(((size_t)b*RR + s)*NPAD + n)*KP + dk] = (__bf16)acc;
  }
}

// ---------------------------------------------------------------------------
// ZtS[b][d][n] = sum_r Zh[b][r][d][n]   (d in 0..100, 100 = D channel)
// ---------------------------------------------------------------------------
__global__ __launch_bounds__(256)
void reduce_kernel(const unsigned short* __restrict__ Zh_us,
                   float* __restrict__ ZtS)
{
  const __bf16* Zh = (const __bf16*)Zh_us;
  const int id = blockIdx.x*256 + threadIdx.x;
  if (id >= BSZ*101*NN) return;
  const int n = id % NN;
  const int d = (id / NN) % 101;
  const int b = id / (NN*101);
  float s = 0.0f;
  for (int rr = 0; rr < RR; ++rr)
    s += (float)Zh[(((size_t)b*RR + rr)*HP + d)*NPAD + n];
  ZtS[((size_t)b*101 + d)*NN + n] = s;
}

// ---------------------------------------------------------------------------
// Zall[b][n][T][e] = Zt/(Dt+EPS); T=0 applies W_pad[0]=w_vs[1][34], T=1: identity.
// ---------------------------------------------------------------------------
template<int T>
__global__ __launch_bounds__(256)
void zall_kernel(const float* __restrict__ ZtS,
                 const float* __restrict__ end_attn,
                 const float* __restrict__ w_vs,
                 float* __restrict__ Zall)
{
  const int id = blockIdx.x*256 + threadIdx.x;
  if (id >= BSZ*NN*HH) return;
  const int e  = id % HH;
  const int t2 = id / HH;
  const int n  = t2 % NN;
  const int b  = t2 / NN;
  const float en = end_attn[b*NN + n];
  const float Dt = ZtS[((size_t)b*101 + 100)*NN + n] * en;
  float z;
  if (T == 0) {
    const float* Wp = w_vs + (size_t)(1*(RR+1) + RR)*(HH*HH);  // w_vs[1][34] = Wi[1]
    z = 0.0f;
    for (int d = 0; d < HH; ++d)
      z += ZtS[((size_t)b*101 + d)*NN + n] * Wp[d*HH + e];
  } else {
    z = ZtS[((size_t)b*101 + e)*NN + n];
  }
  z *= en;
  Zall[(((size_t)b*NN + n)*2 + T)*HH + e] = z / (Dt + 1e-20f);
}

// ---------------------------------------------------------------------------
// Sq[b][e] = S[b] . wq_w[e] + wq_b[e]
// ---------------------------------------------------------------------------
__global__ __launch_bounds__(256)
void sq_kernel(const float* __restrict__ S, const float* __restrict__ wq_w,
               const float* __restrict__ wq_b, float* __restrict__ Sq)
{
  const int id = blockIdx.x*256 + threadIdx.x;
  if (id >= BSZ*HH) return;
  const int e = id % HH;
  const int b = id / HH;
  float s = wq_b[e];
  const float* sr = S + (size_t)b*SD;
  const float* wr = wq_w + (size_t)e*SD;
  for (int d = 0; d < SD; ++d) s += sr[d]*wr[d];
  Sq[b*HH + e] = s;
}

// ---------------------------------------------------------------------------
// attn = softmax_k( Sq . Zall / sqrt(H) );  out = sum_k attn * Zall
// ---------------------------------------------------------------------------
__global__ __launch_bounds__(256)
void attn_out_kernel(const float* __restrict__ Zall,
                     const float* __restrict__ Sq,
                     float* __restrict__ out)
{
  const int id = blockIdx.x*256 + threadIdx.x;
  if (id >= BSZ*NN) return;
  const int n = id % NN;
  const int b = id / NN;
  const float* z0 = Zall + ((size_t)(b*NN + n)*2 + 0)*HH;
  const float* z1 = z0 + HH;
  const float* sq = Sq + b*HH;
  float a0 = 0.0f, a1 = 0.0f;
  for (int e = 0; e < HH; ++e) { a0 += sq[e]*z0[e]; a1 += sq[e]*z1[e]; }
  a0 *= 0.1f; a1 *= 0.1f;   // / sqrt(100)
  const float mx = fmaxf(a0, a1);
  const float e0 = __expf(a0 - mx), e1 = __expf(a1 - mx);
  const float inv = 1.0f/(e0 + e1);
  const float w0 = e0*inv, w1 = e1*inv;
  float* o = out + (size_t)(b*NN + n)*HH;
  for (int e = 0; e < HH; ++e) o[e] = w0*z0[e] + w1*z1[e];
}

// ---------------------------------------------------------------------------
extern "C" void kernel_launch(void* const* d_in, const int* in_sizes, int n_in,
                              void* d_out, int out_size, void* d_ws, size_t ws_size,
                              hipStream_t stream) {
  (void)in_sizes; (void)n_in; (void)out_size; (void)ws_size;
  const float* X     = (const float*)d_in[0];
  const float* A     = (const float*)d_in[1];
  const float* start = (const float*)d_in[2];
  const float* endat = (const float*)d_in[3];
  const float* uni   = (const float*)d_in[4];
  const float* trans = (const float*)d_in[5];
  const float* S     = (const float*)d_in[6];
  const float* w_vs  = (const float*)d_in[7];
  const float* wq_w  = (const float*)d_in[8];
  const float* wq_b  = (const float*)d_in[9];
  float* out = (float*)d_out;

  char* ws = (char*)d_ws;
  size_t off = 0;
  auto take = [&](size_t bytes) -> void* {
    void* p = ws + off;
    off += (bytes + 255) & ~(size_t)255;
    return p;
  };
  unsigned short* Zh   = (unsigned short*)take((size_t)BSZ*RR*HP*NPAD*2);   // ~50.7 MB
  unsigned short* Zmix = (unsigned short*)take((size_t)BSZ*RR*NPAD*KP*2);   // ~57.9 MB
  unsigned short* Ap   = (unsigned short*)take((size_t)BSZ*RR*NPAD*NP*2);   // ~101.4 MB
  unsigned short* Xp   = (unsigned short*)take((size_t)BSZ*NPAD*KP*2);      // ~1.7 MB
  unsigned short* Wp   = (unsigned short*)take((size_t)2*RR*HP*KP*2);       // ~2.0 MB
  float* ZtS  = (float*)take((size_t)BSZ*101*NN*4);                          // ~2.6 MB
  float* Zall = (float*)take((size_t)BSZ*NN*2*HH*4);                         // ~5.1 MB
  float* Sq   = (float*)take((size_t)BSZ*HH*4);

  const dim3 blk(256);

  // pre-pad / convert
  xpad_kernel<<<(BSZ*NPAD*KP + 255)/256, blk, 0, stream>>>(X, start, Xp);
  wpad_kernel<<<(2*RR*HP*KP + 255)/256, blk, 0, stream>>>(w_vs, Wp);
  apad_kernel<<<(BSZ*RR*NPAD*(NP/8) + 255)/256, blk, 0, stream>>>(A, Ap);
  sq_kernel<<<(BSZ*HH + 255)/256, blk, 0, stream>>>(S, wq_w, wq_b, Sq);

  // hop t=0
  stage_kernel<0><<<BSZ*RR, blk, 0, stream>>>(Xp, Ap, Wp, start, uni, Zmix, Zh);
  reduce_kernel<<<(BSZ*101*NN + 255)/256, blk, 0, stream>>>(Zh, ZtS);
  zall_kernel<0><<<(BSZ*NN*HH + 255)/256, blk, 0, stream>>>(ZtS, endat, w_vs, Zall);

  // mix heads (reads t=0 Zh before stage<1> overwrites it)
  mix_kernel<<<(BSZ*NPAD*KP + 255)/256, blk, 0, stream>>>(Zh, trans, Zmix);

  // hop t=1
  stage_kernel<1><<<BSZ*RR, blk, 0, stream>>>(Xp, Ap, Wp, start, uni, Zmix, Zh);
  reduce_kernel<<<(BSZ*101*NN + 255)/256, blk, 0, stream>>>(Zh, ZtS);
  zall_kernel<1><<<(BSZ*NN*HH + 255)/256, blk, 0, stream>>>(ZtS, endat, w_vs, Zall);

  // final attention combine
  attn_out_kernel<<<(BSZ*NN + 255)/256, blk, 0, stream>>>(Zall, Sq, out);
}
